// EEGEmotionGNNSAGE_19628000543387
// MI455X (gfx1250) — compile-verified, matrix operand order checked by reading, not run on hardware
//
#include <hip/hip_runtime.h>

// ---------------------------------------------------------------------------
// GraphSAGE (2 layers, mean agg) + global mean pool + FC for MI455X (gfx1250)
// N=100000 nodes, E=1.6M edges, F=H=128, G=1024 graphs, C=4 classes.
//
// Roofline: edge aggregation is the bound (2 x ~819MB scatter/gather, but x
// and agg both fit in the 192MB L2, so atomics stay on-chip). Dense parts
// (4 x 100Kx128x128 ~ 13 GFLOP) run on v_wmma_f32_16x16x32_bf16 with f32
// accumulation; layer-2 GEMM epilogue fused with graph pooling so h2 is
// never materialized.
// ---------------------------------------------------------------------------

typedef __attribute__((ext_vector_type(16))) __bf16       v16bf;
typedef __attribute__((ext_vector_type(8)))  float        v8f;
typedef __attribute__((ext_vector_type(4)))  unsigned int u32x4;

union Frag { u32x4 q[2]; v16bf v; };

__device__ __forceinline__ unsigned short f2bf(float f) {
    unsigned int u = __builtin_bit_cast(unsigned int, f);
    u += 0x7FFFu + ((u >> 16) & 1u);          // round-to-nearest-even
    return (unsigned short)(u >> 16);
}

__device__ __forceinline__ void atomicAddF(float* p, float v) {
    __hip_atomic_fetch_add(p, v, __ATOMIC_RELAXED, __HIP_MEMORY_SCOPE_AGENT);
}

// ---- degree: deg[dst] += 1 per edge --------------------------------------
__global__ void degree_kernel(const long long* __restrict__ ei, float* __restrict__ deg, int E) {
    int e = blockIdx.x * blockDim.x + threadIdx.x;
    if (e < E) atomicAddF(&deg[(int)ei[(size_t)E + e]], 1.0f);
}

// ---- per-graph node count -------------------------------------------------
__global__ void batch_count_kernel(const long long* __restrict__ batch, float* __restrict__ cnt, int n) {
    int i = blockIdx.x * blockDim.x + threadIdx.x;
    if (i < n) atomicAddF(&cnt[(int)batch[i]], 1.0f);
}

// ---- in-place v = 1 / max(v,1) -------------------------------------------
__global__ void recip_kernel(float* __restrict__ v, int n) {
    int i = blockIdx.x * blockDim.x + threadIdx.x;
    if (i < n) v[i] = 1.0f / fmaxf(v[i], 1.0f);
}

// ---- weights: f32 [k][n] 128x128 -> bf16 transposed [n][k] ---------------
__global__ void wconv_kernel(const float* __restrict__ W, unsigned short* __restrict__ Wt) {
    int i = blockIdx.x * blockDim.x + threadIdx.x;   // 16384 elems
    if (i < 128 * 128) {
        int k = i >> 7, n = i & 127;
        Wt[n * 128 + k] = f2bf(W[i]);
    }
}

// ---- edge scatter: agg[dst][:] += feat[src][:]; one wave32 per edge ------
__global__ __launch_bounds__(256) void scatter_kernel(
        const float* __restrict__ feat, const long long* __restrict__ ei,
        float* __restrict__ agg, int E) {
    int wave = (int)((blockIdx.x * blockDim.x + threadIdx.x) >> 5);
    int lane = threadIdx.x & 31;
    if (wave >= E) return;
    int src = (int)ei[wave];
    int dst = (int)ei[(size_t)E + wave];
    const float* xs = feat + (size_t)src * 128;
    float*       ad = agg  + (size_t)dst * 128;
    __builtin_prefetch(ad + lane, 1, 0);          // global_prefetch_b8
#pragma unroll
    for (int i = 0; i < 4; ++i) {
        int c = lane + 32 * i;
        atomicAddF(ad + c, xs[c]);
    }
}

// ---- fused SAGE layer: out = act( (agg*rdeg)@Wl + xin@Wr + b ) -----------
// 128 threads = 4 waves; block covers 64 rows x 128 cols.
// Each wave: 16-row stripe, 8 col tiles x 4 K-chunks x 2 WMMAs (bf16->f32).
// POOL mode: fuse pooled[batch[row]] += result row (h2 never materialized).
// Epilogue is compile-time specialized; full blocks skip row guards.
template <bool RELU, bool STORE, bool POOL>
__global__ __launch_bounds__(128, 2) void sage_gemm_kernel(
        const float* __restrict__ xin, const float* __restrict__ agg,
        const float* __restrict__ rdeg,
        const unsigned short* __restrict__ Wl_t, const unsigned short* __restrict__ Wr_t,
        const float* __restrict__ bias,
        float* __restrict__ out, float* __restrict__ pooled,
        const long long* __restrict__ batch, int n_rows) {
    __shared__ __align__(16) unsigned short sx[64 * 128];  // xin rows, bf16
    __shared__ __align__(16) unsigned short sm[64 * 128];  // mean rows, bf16

    const int tid  = threadIdx.x;
    const int row0 = blockIdx.x * 64;
    const bool full = (row0 + 64 <= n_rows);

    // Stage 64 rows: convert to bf16, fold in mean = agg * rdeg
    if (full) {
        for (int i = tid; i < 64 * 128; i += 128) {
            int r = i >> 7, c2 = i & 127;
            size_t gidx = (size_t)(row0 + r) * 128 + c2;
            sx[i] = f2bf(xin[gidx]);
            sm[i] = f2bf(agg[gidx] * rdeg[row0 + r]);
        }
    } else {
        for (int i = tid; i < 64 * 128; i += 128) {
            int r = i >> 7, c2 = i & 127;
            int row = row0 + r;
            float xv = 0.f, mv = 0.f;
            if (row < n_rows) {
                xv = xin[(size_t)row * 128 + c2];
                mv = agg[(size_t)row * 128 + c2] * rdeg[row];
            }
            sx[i] = f2bf(xv);
            sm[i] = f2bf(mv);
        }
    }
    __syncthreads();

    const int wave = tid >> 5, lane = tid & 31;
    const int M = lane & 15, half = lane >> 4;
    const int rw = wave * 16;                     // this wave's row stripe

    for (int j = 0; j < 8; ++j) {                 // 8 column tiles of 16
        const int ncol = j * 16 + M;
        const float bval = bias[ncol];
        v8f c;
#pragma unroll
        for (int q = 0; q < 8; ++q) c[q] = bval;  // bias pre-loaded into C

#pragma unroll
        for (int k = 0; k < 4; ++k) {             // K = 128 in chunks of 32
            const int kb = k * 32 + half * 8;     // ISA 16-bit A layout
            Frag am, ax, bl, br;
            const unsigned short* pm = &sm[(rw + M) * 128 + kb];
            am.q[0] = *(const u32x4*)pm;  am.q[1] = *(const u32x4*)(pm + 16);
            const unsigned short* px = &sx[(rw + M) * 128 + kb];
            ax.q[0] = *(const u32x4*)px;  ax.q[1] = *(const u32x4*)(px + 16);
            const unsigned short* pl = &Wl_t[ncol * 128 + kb];
            bl.q[0] = *(const u32x4*)pl;  bl.q[1] = *(const u32x4*)(pl + 16);
            const unsigned short* pr = &Wr_t[ncol * 128 + kb];
            br.q[0] = *(const u32x4*)pr;  br.q[1] = *(const u32x4*)(pr + 16);

            c = __builtin_amdgcn_wmma_f32_16x16x32_bf16(
                    false, am.v, false, bl.v, (short)0, c, false, false);
            c = __builtin_amdgcn_wmma_f32_16x16x32_bf16(
                    false, ax.v, false, br.v, (short)0, c, false, false);
        }

        // Epilogue (compile-time specialized). C layout: M = r + 8*half.
        const int rbase = row0 + rw + 8 * half;
        if (full) {
#pragma unroll
            for (int r = 0; r < 8; ++r) {
                float v = c[r];
                if (RELU) v = fmaxf(v, 0.f);
                int row = rbase + r;
                if (STORE) out[(size_t)row * 128 + ncol] = v;
                if (POOL)  atomicAddF(&pooled[(int)batch[row] * 128 + ncol], v);
            }
        } else {
#pragma unroll
            for (int r = 0; r < 8; ++r) {
                int row = rbase + r;
                if (row < n_rows) {
                    float v = c[r];
                    if (RELU) v = fmaxf(v, 0.f);
                    if (STORE) out[(size_t)row * 128 + ncol] = v;
                    if (POOL)  atomicAddF(&pooled[(int)batch[row] * 128 + ncol], v);
                }
            }
        }
    }
}

// ---- final FC: out[g][c] = (pooled[g]/max(cnt,1)) . Wfc[:,c] + bfc[c] ----
__global__ void fc_kernel(const float* __restrict__ pooled, const float* __restrict__ cnt,
                          const float* __restrict__ Wfc, const float* __restrict__ bfc,
                          float* __restrict__ out, int total) {
    int idx = blockIdx.x * blockDim.x + threadIdx.x;
    if (idx >= total) return;
    int g = idx >> 2, c = idx & 3;
    float acc = 0.f;
    const float* p = pooled + g * 128;
#pragma unroll 8
    for (int h = 0; h < 128; ++h) acc += p[h] * Wfc[h * 4 + c];
    out[idx] = acc / fmaxf(cnt[g], 1.0f) + bfc[c];
}

extern "C" void kernel_launch(void* const* d_in, const int* in_sizes, int n_in,
                              void* d_out, int out_size, void* d_ws, size_t ws_size,
                              hipStream_t stream) {
    const float*     x     = (const float*)d_in[0];
    const long long* ei    = (const long long*)d_in[1];   // (2,E) int64
    const long long* batch = (const long long*)d_in[2];   // (N,)  int64 sorted
    const float*     W1l   = (const float*)d_in[3];
    const float*     W1r   = (const float*)d_in[4];
    const float*     b1    = (const float*)d_in[5];
    const float*     W2l   = (const float*)d_in[6];
    const float*     W2r   = (const float*)d_in[7];
    const float*     b2    = (const float*)d_in[8];
    const float*     Wfc   = (const float*)d_in[9];
    const float*     bfc   = (const float*)d_in[10];
    float*           out   = (float*)d_out;

    const int N = in_sizes[0] / 128;      // 100000
    const int E = in_sizes[1] / 2;        // 1600000
    const int H = 128, C = 4;
    const int G = out_size / C;           // 1024

    // Workspace layout (all float-aligned; wt ends up 16B aligned)
    float* deg    = (float*)d_ws;                          // N   (becomes rdeg)
    float* cnt    = deg + N;                               // G
    float* agg    = cnt + G;                               // N*H (reused both layers)
    float* h1     = agg + (size_t)N * H;                   // N*H
    float* pooled = h1  + (size_t)N * H;                   // G*H
    unsigned short* wt = (unsigned short*)(pooled + (size_t)G * H); // 4 x 128x128 bf16
    unsigned short *wt1l = wt, *wt1r = wt + 16384, *wt2l = wt + 32768, *wt2r = wt + 49152;

    hipMemsetAsync(deg,    0, (size_t)N * sizeof(float), stream);
    hipMemsetAsync(cnt,    0, (size_t)G * sizeof(float), stream);
    hipMemsetAsync(agg,    0, (size_t)N * H * sizeof(float), stream);
    hipMemsetAsync(pooled, 0, (size_t)G * H * sizeof(float), stream);

    degree_kernel     <<<(E + 255) / 256, 256, 0, stream>>>(ei, deg, E);
    batch_count_kernel<<<(N + 255) / 256, 256, 0, stream>>>(batch, cnt, N);
    recip_kernel      <<<(N + 255) / 256, 256, 0, stream>>>(deg, N);

    wconv_kernel<<<64, 256, 0, stream>>>(W1l, wt1l);
    wconv_kernel<<<64, 256, 0, stream>>>(W1r, wt1r);
    wconv_kernel<<<64, 256, 0, stream>>>(W2l, wt2l);
    wconv_kernel<<<64, 256, 0, stream>>>(W2r, wt2r);

    const int gemm_blocks = (N + 63) / 64;

    // Layer 1: scatter x, then h1 = relu(mean@W1l + x@W1r + b1)
    scatter_kernel<<<(E + 7) / 8, 256, 0, stream>>>(x, ei, agg, E);
    sage_gemm_kernel<true, true, false><<<gemm_blocks, 128, 0, stream>>>(
        x, agg, deg, wt1l, wt1r, b1, h1, nullptr, batch, N);

    // Layer 2: scatter h1, then h2 = mean@W2l + h1@W2r + b2, fused pooling
    hipMemsetAsync(agg, 0, (size_t)N * H * sizeof(float), stream);
    scatter_kernel<<<(E + 7) / 8, 256, 0, stream>>>(h1, ei, agg, E);
    sage_gemm_kernel<false, false, true><<<gemm_blocks, 128, 0, stream>>>(
        h1, agg, deg, wt2l, wt2r, b2, nullptr, pooled, batch, N);

    // Final FC over pooled means
    fc_kernel<<<(G * C + 255) / 256, 256, 0, stream>>>(pooled, cnt, Wfc, bfc, out, G * C);
}
